// AttentionModule_30021821399395
// MI455X (gfx1250) — matrix operand order repaired and checked by reading, not compiled, not run on hardware
//
#include <hip/hip_runtime.h>

typedef __attribute__((ext_vector_type(16))) _Float16 v16h;
typedef __attribute__((ext_vector_type(8)))  _Float16 v8h;
typedef __attribute__((ext_vector_type(4)))  _Float16 v4h;
typedef __attribute__((ext_vector_type(8)))  float    v8f;
typedef __attribute__((ext_vector_type(4)))  float    v4f;

#define BB   4
#define SS   2048
#define DD   1024
#define HH   16
#define CC   64
#define HID  1024
#define MTOK (BB * SS)   /* 8192 tokens */
#define NQKV (3 * HID)   /* 3072 fused qkv columns */
#define BK   32          /* k-step per LDS tile */
#define KDEPTH (DD / BK) /* 32 k-steps */

// ---------------------------------------------------------------------------
// CDNA5 async global->LDS copy (ASYNCcnt-tracked DMA, ISA 10.x / 08_async).
// vdst = LDS byte address VGPR, vaddr = 64-bit global address.
// ---------------------------------------------------------------------------
__device__ __forceinline__ void async_copy_b128(unsigned lds_off, const void* gaddr) {
  asm volatile("global_load_async_to_lds_b128 %0, %1, off"
               :: "v"(lds_off), "v"(gaddr) : "memory");
}

template <int N>
__device__ __forceinline__ void wait_asynccnt() {
#if __has_builtin(__builtin_amdgcn_s_wait_asynccnt)
  __builtin_amdgcn_s_wait_asynccnt(N);
#else
  asm volatile("s_wait_asynccnt %0" :: "i"(N) : "memory");
#endif
}

// ---------------------------------------------------------------------------
// Stage 0a: elementwise f32 -> f16 conversion (x)
// ---------------------------------------------------------------------------
__global__ void cvt_f32_f16(const float* __restrict__ src,
                            _Float16* __restrict__ dst, int n) {
  int i = (blockIdx.x * blockDim.x + threadIdx.x) * 4;
  if (i + 3 < n) {
    v4f x = *reinterpret_cast<const v4f*>(src + i);
    v4h o;
    o[0] = (_Float16)x[0]; o[1] = (_Float16)x[1];
    o[2] = (_Float16)x[2]; o[3] = (_Float16)x[3];
    *reinterpret_cast<v4h*>(dst + i) = o;
  }
}

// ---------------------------------------------------------------------------
// Stage 0b: tiled transpose + f32->f16 convert.  W[K][N] -> Wt[N][K]
// ---------------------------------------------------------------------------
__global__ void transpose_cvt(const float* __restrict__ W,
                              _Float16* __restrict__ Wt, int K, int N) {
  __shared__ float tile[32][33];
  int tx = threadIdx.x;              // 0..31
  int ty = threadIdx.y;              // 0..7
  int k0 = blockIdx.y * 32;
  int n0 = blockIdx.x * 32;
#pragma unroll
  for (int i = 0; i < 32; i += 8)
    tile[ty + i][tx] = W[(size_t)(k0 + ty + i) * N + n0 + tx];
  __syncthreads();
#pragma unroll
  for (int i = 0; i < 32; i += 8)
    Wt[(size_t)(n0 + ty + i) * K + k0 + tx] = (_Float16)tile[tx][ty + i];
}

// ---------------------------------------------------------------------------
// LDS-staged, async double-buffered WMMA GEMM.
//   Block = 128 threads = 4 waves, 128x128 output tile, wave -> 64x64 subtile.
//   Per k-step: A tile 128xBK and N-major B tile 128xBK staged in LDS via
//   global_load_async_to_lds_b128 (8 async instrs per wave per buffer, so
//   s_wait_asynccnt 8 retires exactly the previous buffer, in-order queue).
//   Fragment layouts (contiguous k, same order for A and B):
//     A (16-bit 16x32): lane L -> row L%16, K = 16*(L/16) + 0..15
//     B (32x16):        lane L -> col L%16, K = 16*(L/16) + 0..15
//   MODE 0: fused QKV epilogue (bias + fold 1/sqrt(C) into Q, f16 out)
//   MODE 1: output projection epilogue (bias, f32 out)
// ---------------------------------------------------------------------------
template <int MODE>
__global__ __launch_bounds__(128) void gemm_wmma(
    const _Float16* __restrict__ A, const _Float16* __restrict__ Bt,
    const float* __restrict__ bias0, const float* __restrict__ bias1,
    const float* __restrict__ bias2,
    _Float16* __restrict__ outh, float* __restrict__ outf, int ldo) {
  __shared__ _Float16 smem[2][2][128][BK];   // [buf][A|B][row][k]  = 32 KB
  const int t    = threadIdx.x;              // 0..127
  const int lane = t & 31;
  const int w    = t >> 5;
  const int gn0  = blockIdx.x * 128;
  const int gm0  = blockIdx.y * 128;
  const int sm0  = (w >> 1) * 64;
  const int sn0  = (w & 1) * 64;
  const int hi   = lane >> 4;
  const int lq   = lane & 15;

  // thread t owns row t of the A tile and row t of the (N-major) B tile
  const _Float16* gA = A  + (size_t)(gm0 + t) * DD;
  const _Float16* gB = Bt + (size_t)(gn0 + t) * DD;

  auto issue = [&](int buf, int k0) {
    unsigned la = (unsigned)(unsigned long long)(&smem[buf][0][t][0]);
    unsigned lb = (unsigned)(unsigned long long)(&smem[buf][1][t][0]);
#pragma unroll
    for (int s = 0; s < 4; ++s) {
      async_copy_b128(la + s * 16, gA + k0 + s * 8);
      async_copy_b128(lb + s * 16, gB + k0 + s * 8);
    }
  };

  v8f acc[4][4] = {};
  issue(0, 0);
  int buf = 0;
  for (int kt = 0; kt < KDEPTH; ++kt) {
    if (kt + 1 < KDEPTH) {
      issue(buf ^ 1, (kt + 1) * BK);
      wait_asynccnt<8>();   // previous buffer's 8 async copies are done
    } else {
      wait_asynccnt<0>();
    }
    __syncthreads();

    v16h a[4], b[4];
#pragma unroll
    for (int i = 0; i < 4; ++i) {
      v8h alo = *(const v8h*)&smem[buf][0][sm0 + i * 16 + lq][hi * 16];
      v8h ahi = *(const v8h*)&smem[buf][0][sm0 + i * 16 + lq][hi * 16 + 8];
#pragma unroll
      for (int e = 0; e < 8; ++e) { a[i][e] = alo[e]; a[i][e + 8] = ahi[e]; }
      b[i] = *(const v16h*)&smem[buf][1][sn0 + i * 16 + lq][hi * 16];
    }
#pragma unroll
    for (int mf = 0; mf < 4; ++mf)
#pragma unroll
      for (int nf = 0; nf < 4; ++nf)
        acc[mf][nf] = __builtin_amdgcn_wmma_f32_16x16x32_f16(
            false, a[mf], false, b[nf], (short)0, acc[mf][nf], false, false);

    __syncthreads();
    buf ^= 1;
  }

  // ---- epilogue -----------------------------------------------------------
#pragma unroll
  for (int nf = 0; nf < 4; ++nf) {
    const int n = gn0 + sn0 + nf * 16 + lq;
    float bias, scale = 1.0f;
    if (MODE == 0) {
      const int which = n >> 10;          // 0=Q 1=K 2=V
      const int nn = n & (HID - 1);
      bias = (which == 0) ? bias0[nn] : ((which == 1) ? bias1[nn] : bias2[nn]);
      if (which == 0) scale = 0.125f;     // 1/sqrt(64), applied after bias
    } else {
      bias = bias0[n];
    }
#pragma unroll
    for (int mf = 0; mf < 4; ++mf) {
#pragma unroll
      for (int v = 0; v < 8; ++v) {
        const int m = gm0 + sm0 + mf * 16 + hi * 8 + v;
        const float cv = (acc[mf][nf][v] + bias) * scale;
        if (MODE == 0) outh[(size_t)m * ldo + n] = (_Float16)cv;
        else           outf[(size_t)m * ldo + n] = cv;
      }
    }
  }
}

// ---------------------------------------------------------------------------
// Stage 2: per-(token,head) channel attention (fused: never materialize the
// 2.1 GB scores tensor).  scores[i][j] = q[i]*k[j], causal mask j<=i, softmax
// over j, attn[i] = sum_j w[i][j]*v[j].  One wave per (token,head).
// ---------------------------------------------------------------------------
__global__ void chan_attn(const _Float16* __restrict__ qkv16,
                          _Float16* __restrict__ attn16) {
  __shared__ float qs[8][CC], ks[8][CC], vs[8][CC];
  const int lane = threadIdx.x & 31;
  const int w = threadIdx.x >> 5;
  const int g = blockIdx.x * 8 + w;
  const int token = g >> 4;
  const int head = g & 15;
  const _Float16* base = qkv16 + (size_t)token * NQKV + head * CC;
#pragma unroll
  for (int t = lane; t < CC; t += 32) {
    qs[w][t] = (float)base[t];
    ks[w][t] = (float)base[HID + t];
    vs[w][t] = (float)base[2 * HID + t];
  }
  __syncthreads();
  _Float16* orow = attn16 + (size_t)token * HID + head * CC;
  for (int r = lane; r < CC; r += 32) {
    const float qi = qs[w][r];
    float m = -INFINITY;
    for (int j = 0; j <= r; ++j) m = fmaxf(m, qi * ks[w][j]);
    float sum = 0.0f, acc = 0.0f;
    for (int j = 0; j <= r; ++j) {
      const float e = __expf(qi * ks[w][j] - m);
      sum += e;
      acc += e * vs[w][j];
    }
    orow[r] = (_Float16)(acc / sum);
  }
}

// ---------------------------------------------------------------------------
extern "C" void kernel_launch(void* const* d_in, const int* in_sizes, int n_in,
                              void* d_out, int out_size, void* d_ws, size_t ws_size,
                              hipStream_t stream) {
  const float* x  = (const float*)d_in[0];
  const float* Wq = (const float*)d_in[1];
  const float* bq = (const float*)d_in[2];
  const float* Wk = (const float*)d_in[3];
  const float* bk = (const float*)d_in[4];
  const float* Wv = (const float*)d_in[5];
  const float* bv = (const float*)d_in[6];
  const float* Wo = (const float*)d_in[7];
  const float* bo = (const float*)d_in[8];
  float* out = (float*)d_out;

  char* ws = (char*)d_ws;
  _Float16* x16    = (_Float16*)(ws);                         // 16 MB
  _Float16* Wt     = (_Float16*)(ws + (16ull << 20));         //  6 MB [3072][1024]
  _Float16* Wot    = (_Float16*)(ws + (22ull << 20));         //  2 MB [1024][1024]
  _Float16* qkv16  = (_Float16*)(ws + (24ull << 20));         // 48 MB [8192][3072]
  _Float16* attn16 = (_Float16*)(ws + (72ull << 20));         // 16 MB [8192][1024]

  // Stage 0: precision conversion + weight transposes
  {
    const int n = MTOK * DD;
    cvt_f32_f16<<<n / (256 * 4), 256, 0, stream>>>(x, x16, n);
  }
  dim3 tb(32, 8);
  dim3 tg(HID / 32, DD / 32);
  transpose_cvt<<<tg, tb, 0, stream>>>(Wq, Wt,                         DD, HID);
  transpose_cvt<<<tg, tb, 0, stream>>>(Wk, Wt + (size_t)HID * DD,      DD, HID);
  transpose_cvt<<<tg, tb, 0, stream>>>(Wv, Wt + (size_t)2 * HID * DD,  DD, HID);
  transpose_cvt<<<dim3(DD / 32, HID / 32), tb, 0, stream>>>(Wo, Wot, HID, DD);

  // Stage 1: fused QKV projection (async-LDS staged WMMA)
  gemm_wmma<0><<<dim3(NQKV / 128, MTOK / 128), 128, 0, stream>>>(
      x16, Wt, bq, bk, bv, qkv16, nullptr, NQKV);

  // Stage 2: fused per-token channel attention
  chan_attn<<<(MTOK * HH) / 8, 256, 0, stream>>>(qkv16, attn16);

  // Stage 3: output projection (async-LDS staged WMMA) + bias, fp32 out
  gemm_wmma<1><<<dim3(DD / 128, MTOK / 128), 128, 0, stream>>>(
      attn16, Wot, bo, nullptr, nullptr, nullptr, out, DD);
}
